// MultiHeadSelfAttention_3788161155075
// MI455X (gfx1250) — matrix operand order repaired and checked by reading, not compiled, not run on hardware
//
#include <hip/hip_runtime.h>
#include <hip/hip_bf16.h>
#include <stdint.h>

// ---------------------------------------------------------------------------
// CDNA5 (gfx1250) fused multi-head self-attention.
// bf16 WMMA (f32 accumulate) + async global->LDS DMA with double buffering.
// ---------------------------------------------------------------------------

typedef __attribute__((ext_vector_type(16))) __bf16 v16bf;
typedef __attribute__((ext_vector_type(8)))  float  v8f;

union FragAB { v16bf v; uint32_t u[8]; unsigned short h[16]; };
union FragC  { v8f v; float f[8]; };

__device__ __forceinline__ unsigned short f2bf(float x) {
    union { float f; uint32_t u; } c; c.f = x;
    uint32_t r = (c.u + 0x7FFFu + ((c.u >> 16) & 1u)) >> 16;  // RNE
    return (unsigned short)r;
}

__device__ __forceinline__ uint32_t pack2bf(float a, float b) {
#if __has_builtin(__builtin_amdgcn_cvt_pk_bf16_f32)
    typedef __attribute__((ext_vector_type(2))) __bf16 v2bf;
    union { v2bf v; uint32_t u; } cv;
    cv.v = __builtin_amdgcn_cvt_pk_bf16_f32(a, b);
    return cv.u;
#else
    return (uint32_t)f2bf(a) | ((uint32_t)f2bf(b) << 16);
#endif
}

__device__ __forceinline__ v8f wmma_bf16(const v16bf a, const v16bf b, const v8f c) {
    // D = A(16x32 bf16) * B(32x16 bf16) + C(16x16 f32)
    return __builtin_amdgcn_wmma_f32_16x16x32_bf16(
        /*neg_a=*/false, a, /*neg_b=*/false, b,
        /*c_mod=*/(short)0, c, /*reuse_a=*/false, /*reuse_b=*/false);
}

// Async DMA: 16 bytes per lane, global -> LDS, tracked by ASYNCcnt.
// Flat shared-pointer low 32 bits == wave-relative LDS byte address (ISA 10.2).
__device__ __forceinline__ void async_copy_b128(void* lds_dst, const void* src) {
    const uint32_t lds = (uint32_t)(uintptr_t)lds_dst;
    asm volatile("global_load_async_to_lds_b128 %0, %1, off"
                 :: "v"(lds), "v"(src)
                 : "memory");
}
__device__ __forceinline__ void wait_async0() {
    asm volatile("s_wait_asynccnt 0x0" ::: "memory");
}

#define DMODEL 256
#define SEQ    1024
#define NHEAD  8
#define DK     32
#define NROWS  32768   // B*S

// ---------------------------------------------------------------------------
// Kernel 0: one-shot f32 -> bf16 conversion (X and weight matrices).
// ---------------------------------------------------------------------------
__global__ __launch_bounds__(256) void cvt_f32_bf16(
    const float* __restrict__ src, unsigned short* __restrict__ dst, int n4)
{
    const int stride = gridDim.x * blockDim.x;
    for (int i = blockIdx.x * blockDim.x + threadIdx.x; i < n4; i += stride) {
        const float4 v = *(const float4*)&src[(size_t)i * 4];
        uint2 p; p.x = pack2bf(v.x, v.y); p.y = pack2bf(v.z, v.w);
        *(uint2*)&dst[(size_t)i * 4] = p;
    }
}

// ---------------------------------------------------------------------------
// Shared staging helper: async-issue one K-stage (A 64x32, B 256x32 bf16).
// 5 async b128 chunks per thread (1 A + 4 B).
// ---------------------------------------------------------------------------
__device__ __forceinline__ void stage_async(
    unsigned short (*A)[32], unsigned short (*B)[32],
    const unsigned short* __restrict__ Abf, const unsigned short* __restrict__ Wbf,
    int m0, int ko, int tid)
{
    const int ar = tid >> 2, ac = (tid & 3) * 8;
    async_copy_b128(&A[ar][ac], &Abf[(size_t)(m0 + ar) * DMODEL + ko + ac]);
    #pragma unroll
    for (int j = 0; j < 4; ++j) {
        const int i = tid + j * 256;
        const int n = i >> 2, c8 = (i & 3) * 8;
        async_copy_b128(&B[n][c8], &Wbf[(size_t)n * DMODEL + ko + c8]);
    }
}

// ---------------------------------------------------------------------------
// Kernel 1: projection  dst = bf16( Xbf @ Wbf^T + bias ), head-layout scatter.
// Block: 256 threads (8 waves). Each block: 64 rows x full 256 cols.
// transposed==0: dst[((b*8+h)*1024+s)*32 + d]   (Q, K: [bh][s][d])
// transposed==1: dst[((b*8+h)*32+d)*1024 + s]   (V:   [bh][d][s])
// ---------------------------------------------------------------------------
__global__ __launch_bounds__(256) void mhsa_qkv_proj(
    const unsigned short* __restrict__ Xbf,   // [32768][256] bf16
    const unsigned short* __restrict__ Wbf,   // [256][256]   bf16
    const float* __restrict__ bias, unsigned short* __restrict__ dst,
    int transposed)
{
    __shared__ unsigned short As[2][64][32];
    __shared__ unsigned short Bs[2][256][32];

    const int tid  = threadIdx.x;
    const int wave = tid >> 5;
    const int lane = tid & 31;
    const int lh   = lane >> 4;     // lane-half (0/1)
    const int ln   = lane & 15;
    const int m0   = blockIdx.x * 64;

    FragC c[4][2];
    for (int mt = 0; mt < 4; ++mt)
        for (int nt = 0; nt < 2; ++nt)
            for (int r = 0; r < 8; ++r) c[mt][nt].f[r] = 0.f;

    stage_async(As[0], Bs[0], Xbf, Wbf, m0, 0, tid);

    #pragma unroll
    for (int it = 0; it < 8; ++it) {          // K stages of 32
        const int cur = it & 1;
        wait_async0();
        __syncthreads();
        if (it < 7) stage_async(As[cur ^ 1], Bs[cur ^ 1], Xbf, Wbf, m0, (it + 1) * 32, tid);

        // A fragments: lanes 0-15 hold K 0..7 / 16..23, lanes 16-31 hold 8..15 / 24..31
        FragAB a[4];
        for (int mt = 0; mt < 4; ++mt) {
            const uint32_t* r32 = (const uint32_t*)&As[cur][mt * 16 + ln][0];
            const int base = lh * 4;
            #pragma unroll
            for (int v = 0; v < 4; ++v) { a[mt].u[v] = r32[base + v]; a[mt].u[4 + v] = r32[8 + base + v]; }
        }
        // B fragments + 8 WMMAs
        for (int ntl = 0; ntl < 2; ++ntl) {
            const int n = (wave * 2 + ntl) * 16 + ln;
            FragAB b;
            const uint32_t* r32 = (const uint32_t*)&Bs[cur][n][0];
            const int base = lh * 4;   // lanes 0-15: K 0..7 / 16..23, lanes 16-31: K 8..15 / 24..31
            #pragma unroll
            for (int v = 0; v < 4; ++v) { b.u[v] = r32[base + v]; b.u[4 + v] = r32[8 + base + v]; }
            #pragma unroll
            for (int mt = 0; mt < 4; ++mt)
                c[mt][ntl].v = wmma_bf16(a[mt].v, b.v, c[mt][ntl].v);
        }
    }

    // Epilogue: bias add, bf16 convert, head scatter
    for (int mt = 0; mt < 4; ++mt)
        for (int ntl = 0; ntl < 2; ++ntl) {
            const int col = (wave * 2 + ntl) * 16 + ln;
            const float bv = bias[col];
            const int h = col >> 5, d = col & 31;
            #pragma unroll
            for (int r = 0; r < 8; ++r) {
                const int row = m0 + mt * 16 + lh * 8 + r;
                const int b = row >> 10, s = row & 1023;
                const unsigned short val = f2bf(c[mt][ntl].f[r] + bv);
                size_t idx;
                if (transposed) idx = ((size_t)(b * NHEAD + h) * DK + d) * SEQ + s;
                else            idx = ((size_t)(b * NHEAD + h) * SEQ + s) * DK + d;
                dst[idx] = val;
            }
        }
}

// ---------------------------------------------------------------------------
// Kernel 2: fused attention. One block per (bh, 16-query tile). 128 thr = 4 waves.
// Dynamic LDS: sc[16][1024] f32 scores | red[16][8] | part[4][16][32]
// ---------------------------------------------------------------------------
#define ATTN_LDS_BYTES (16 * 1024 * 4 + 16 * 8 * 4 + 4 * 16 * 32 * 4)

__global__ __launch_bounds__(128) void mhsa_attention(
    const unsigned short* __restrict__ Qbf,   // [bh][s][d]
    const unsigned short* __restrict__ Kbf,   // [bh][s][d]
    const unsigned short* __restrict__ Vt,    // [bh][d][s]
    const unsigned char*  __restrict__ maskp, // [S][S] bool
    unsigned short* __restrict__ O)           // [B*S][256] bf16
{
    extern __shared__ char smem[];
    float* sc   = (float*)smem;                       // 16 x 1024
    float* red  = (float*)(smem + 16 * 1024 * 4);     // 16 x 8
    float* part = red + 16 * 8;                       // 4 x 16 x 32

    const int bh = blockIdx.y, qt = blockIdx.x;
    const unsigned short* Qh = Qbf + (size_t)bh * SEQ * DK;
    const unsigned short* Kh = Kbf + (size_t)bh * SEQ * DK;
    const unsigned short* Vh = Vt  + (size_t)bh * DK * SEQ;

    const int tid = threadIdx.x;
    const int wave = tid >> 5, lane = tid & 31;
    const int lh = lane >> 4, ln = lane & 15;

    // --- Phase 1: scores = scale * Q Kt into LDS (mask applied in phase 2) -
    FragAB aq;
    {
        const uint32_t* qrow = (const uint32_t*)(Qh + (size_t)(qt * 16 + ln) * DK);
        const int base = lh * 4;
        #pragma unroll
        for (int v = 0; v < 4; ++v) { aq.u[v] = qrow[base + v]; aq.u[4 + v] = qrow[8 + base + v]; }
    }
    const float scale = 0.17677669529663687f;   // 1/sqrt(32)
    for (int kt = wave * 16; kt < wave * 16 + 16; ++kt) {
        FragAB bk;
        const uint32_t* krow = (const uint32_t*)(Kh + (size_t)(kt * 16 + ln) * DK);
        const int base = lh * 4;
        #pragma unroll
        for (int v = 0; v < 4; ++v) { bk.u[v] = krow[base + v]; bk.u[4 + v] = krow[8 + base + v]; }
        FragC s;
        #pragma unroll
        for (int r = 0; r < 8; ++r) s.f[r] = 0.f;
        s.v = wmma_bf16(aq.v, bk.v, s.v);
        const int key = kt * 16 + ln;
        #pragma unroll
        for (int r = 0; r < 8; ++r) {
            const int m = lh * 8 + r;
            sc[m * SEQ + key] = s.f[r] * scale;
        }
    }
    __syncthreads();

    // --- Phase 2: mask + row softmax (8 threads/row, 128 contiguous cols) --
    const int row = tid >> 3, seg = tid & 7;
    float* srow = sc + row * SEQ;
    const uint32_t* mrow = (const uint32_t*)(maskp + (size_t)(qt * 16 + row) * SEQ);
    float mx = -3.4e38f;
    for (int j4 = seg * 32; j4 < seg * 32 + 32; ++j4) {   // 32 dwords = 128 bools
        const uint32_t m4 = mrow[j4];
        #pragma unroll
        for (int e = 0; e < 4; ++e) {
            float v = srow[j4 * 4 + e];
            if (((m4 >> (8 * e)) & 0xffu) == 0u) v = -1e9f;
            srow[j4 * 4 + e] = v;
            mx = fmaxf(mx, v);
        }
    }
    red[row * 8 + seg] = mx;
    __syncthreads();
    float rmx = red[row * 8];
    #pragma unroll
    for (int j = 1; j < 8; ++j) rmx = fmaxf(rmx, red[row * 8 + j]);
    __syncthreads();
    float sum = 0.f;
    for (int j = seg * 128; j < seg * 128 + 128; ++j) {
        const float e = __expf(srow[j] - rmx);
        srow[j] = e; sum += e;
    }
    red[row * 8 + seg] = sum;
    __syncthreads();
    float rsum = 0.f;
    #pragma unroll
    for (int j = 0; j < 8; ++j) rsum += red[row * 8 + j];
    const float inv = 1.0f / rsum;
    for (int j = seg * 128; j < seg * 128 + 128; ++j) srow[j] *= inv;
    __syncthreads();

    // --- Phase 3: out = P @ V (per-wave K-split over 8 kc chunks of 32) ----
    FragC o0, o1;
    #pragma unroll
    for (int r = 0; r < 8; ++r) { o0.f[r] = 0.f; o1.f[r] = 0.f; }
    const uint32_t* v0r = (const uint32_t*)(Vh + (size_t)ln * SEQ);
    const uint32_t* v1r = (const uint32_t*)(Vh + (size_t)(16 + ln) * SEQ);
    for (int kc = wave * 8; kc < wave * 8 + 8; ++kc) {
        FragAB ap;
        const float* pr = sc + ln * SEQ + kc * 32;
        #pragma unroll
        for (int v = 0; v < 8; ++v) {
            const int k0 = ((v >> 2) << 4) + (lh << 3) + ((v & 3) << 1);
            ap.u[v] = pack2bf(pr[k0], pr[k0 + 1]);
        }
        FragAB b0, b1;
        const int boff = kc * 16 + lh * 4;   // dword offset into Vt row
        #pragma unroll
        for (int v = 0; v < 4; ++v) { b0.u[v] = v0r[boff + v]; b1.u[v] = v1r[boff + v]; b0.u[4 + v] = v0r[boff + 8 + v]; b1.u[4 + v] = v1r[boff + 8 + v]; }
        o0.v = wmma_bf16(ap.v, b0.v, o0.v);
        o1.v = wmma_bf16(ap.v, b1.v, o1.v);
    }
    // Per-wave partials -> LDS
    float* pw = part + wave * 16 * 32;
    #pragma unroll
    for (int r = 0; r < 8; ++r) {
        const int m = lh * 8 + r;
        pw[m * 32 + ln]      = o0.f[r];
        pw[m * 32 + 16 + ln] = o1.f[r];
    }
    __syncthreads();
    // Reduce 4 partials, store packed bf16 pairs to O row-major [B*S][256]
    const int b = bh >> 3, h = bh & 7;
    for (int i = tid; i < 16 * 16; i += 128) {      // 256 dword stores
        const int m = i >> 4, d = (i & 15) * 2;
        const int e0 = m * 32 + d;
        const float a0 = part[e0]     + part[512 + e0]     + part[1024 + e0]     + part[1536 + e0];
        const float a1 = part[e0 + 1] + part[512 + e0 + 1] + part[1024 + e0 + 1] + part[1536 + e0 + 1];
        const size_t orow = (size_t)b * SEQ + qt * 16 + m;
        *(uint32_t*)&O[orow * DMODEL + h * DK + d] = pack2bf(a0, a1);
    }
}

// ---------------------------------------------------------------------------
// Kernel 3: output projection  Y = O @ Wo^T + bo  (fp32 out)
// ---------------------------------------------------------------------------
__global__ __launch_bounds__(256) void mhsa_out_proj(
    const unsigned short* __restrict__ O, const unsigned short* __restrict__ Wbf,
    const float* __restrict__ bias, float* __restrict__ Y)
{
    __shared__ unsigned short As[2][64][32];
    __shared__ unsigned short Bs[2][256][32];

    const int tid  = threadIdx.x;
    const int wave = tid >> 5;
    const int lane = tid & 31;
    const int lh   = lane >> 4;
    const int ln   = lane & 15;
    const int m0   = blockIdx.x * 64;

    FragC c[4][2];
    for (int mt = 0; mt < 4; ++mt)
        for (int nt = 0; nt < 2; ++nt)
            for (int r = 0; r < 8; ++r) c[mt][nt].f[r] = 0.f;

    stage_async(As[0], Bs[0], O, Wbf, m0, 0, tid);

    #pragma unroll
    for (int it = 0; it < 8; ++it) {
        const int cur = it & 1;
        wait_async0();
        __syncthreads();
        if (it < 7) stage_async(As[cur ^ 1], Bs[cur ^ 1], O, Wbf, m0, (it + 1) * 32, tid);

        FragAB a[4];
        for (int mt = 0; mt < 4; ++mt) {
            const uint32_t* r32 = (const uint32_t*)&As[cur][mt * 16 + ln][0];
            const int base = lh * 4;
            #pragma unroll
            for (int v = 0; v < 4; ++v) { a[mt].u[v] = r32[base + v]; a[mt].u[4 + v] = r32[8 + base + v]; }
        }
        for (int ntl = 0; ntl < 2; ++ntl) {
            const int n = (wave * 2 + ntl) * 16 + ln;
            FragAB b;
            const uint32_t* r32 = (const uint32_t*)&Bs[cur][n][0];
            const int base = lh * 4;
            #pragma unroll
            for (int v = 0; v < 4; ++v) { b.u[v] = r32[base + v]; b.u[4 + v] = r32[8 + base + v]; }
            #pragma unroll
            for (int mt = 0; mt < 4; ++mt)
                c[mt][ntl].v = wmma_bf16(a[mt].v, b.v, c[mt][ntl].v);
        }
    }

    for (int mt = 0; mt < 4; ++mt)
        for (int ntl = 0; ntl < 2; ++ntl) {
            const int col = (wave * 2 + ntl) * 16 + ln;
            const float bv = bias[col];
            #pragma unroll
            for (int r = 0; r < 8; ++r) {
                const int rowg = m0 + mt * 16 + lh * 8 + r;
                Y[(size_t)rowg * DMODEL + col] = c[mt][ntl].f[r] + bv;
            }
        }
}

// ---------------------------------------------------------------------------
extern "C" void kernel_launch(void* const* d_in, const int* in_sizes, int n_in,
                              void* d_out, int out_size, void* d_ws, size_t ws_size,
                              hipStream_t stream) {
    const float*         x    = (const float*)d_in[0];
    const unsigned char* mask = (const unsigned char*)d_in[1];
    const float* Wq = (const float*)d_in[2];
    const float* bq = (const float*)d_in[3];
    const float* Wk = (const float*)d_in[4];
    const float* bk = (const float*)d_in[5];
    const float* Wv = (const float*)d_in[6];
    const float* bv = (const float*)d_in[7];
    const float* Wo = (const float*)d_in[8];
    const float* bo = (const float*)d_in[9];

    const size_t headElems = (size_t)32 * NHEAD * SEQ * DK;   // 8,388,608
    const size_t xElems    = (size_t)NROWS * DMODEL;          // 8,388,608
    const size_t wElems    = (size_t)DMODEL * DMODEL;         // 65,536

    unsigned short* Q   = (unsigned short*)d_ws;
    unsigned short* K   = Q   + headElems;
    unsigned short* V   = K   + headElems;
    unsigned short* O   = V   + headElems;
    unsigned short* Xbf = O   + headElems;
    unsigned short* Wqb = Xbf + xElems;
    unsigned short* Wkb = Wqb + wElems;
    unsigned short* Wvb = Wkb + wElems;
    unsigned short* Wob = Wvb + wElems;

    // One-shot f32 -> bf16 conversions (X and the four weight matrices)
    cvt_f32_bf16<<<dim3(2048), dim3(256), 0, stream>>>(x,  Xbf, (int)(xElems / 4));
    cvt_f32_bf16<<<dim3(64),   dim3(256), 0, stream>>>(Wq, Wqb, (int)(wElems / 4));
    cvt_f32_bf16<<<dim3(64),   dim3(256), 0, stream>>>(Wk, Wkb, (int)(wElems / 4));
    cvt_f32_bf16<<<dim3(64),   dim3(256), 0, stream>>>(Wv, Wvb, (int)(wElems / 4));
    cvt_f32_bf16<<<dim3(64),   dim3(256), 0, stream>>>(Wo, Wob, (int)(wElems / 4));

    // Projections: 512 blocks x 256 threads each
    mhsa_qkv_proj<<<dim3(NROWS / 64), dim3(256), 0, stream>>>(Xbf, Wqb, bq, Q, 0);
    mhsa_qkv_proj<<<dim3(NROWS / 64), dim3(256), 0, stream>>>(Xbf, Wkb, bk, K, 0);
    mhsa_qkv_proj<<<dim3(NROWS / 64), dim3(256), 0, stream>>>(Xbf, Wvb, bv, V, 1);

    // Fused attention: (S/16) x (B*H) blocks, 128 threads, ~74 KB dynamic LDS
    (void)hipFuncSetAttribute((const void*)mhsa_attention,
                              hipFuncAttributeMaxDynamicSharedMemorySize,
                              ATTN_LDS_BYTES);
    mhsa_attention<<<dim3(SEQ / 16, 32 * NHEAD), dim3(128), ATTN_LDS_BYTES, stream>>>(
        Q, K, V, mask, O);

    // Output projection (fp32 result)
    mhsa_out_proj<<<dim3(NROWS / 64), dim3(256), 0, stream>>>(O, Wob, bo, (float*)d_out);
}
